// RGCN_56788057588583
// MI455X (gfx1250) — matrix operand order repaired and checked by reading, not compiled, not run on hardware
//
#include <hip/hip_runtime.h>
#include <hip/hip_bf16.h>

typedef __attribute__((ext_vector_type(16))) _Float16 v16h;
typedef __attribute__((ext_vector_type(8)))  float    v8f;

#define NUM_RELS 16
#define DIN 64

union F16x16 { v16h v; _Float16 h[16]; float4 q[2]; };

// A fragment (16x32 f16, M=lane&15). ISA layout: lanes 0-15: K=j (j<8), K=8+j (j>=8);
// lanes 16-31: K=8+j (j<8), K=16+j (j>=8). Two contiguous 16B chunks per lane.
__device__ __forceinline__ v16h loadA(const _Float16* __restrict__ row, int lane, int kb) {
    const int hi = (lane & 16) ? 8 : 0;
    F16x16 u;
    u.q[0] = *(const float4*)(row + kb + hi);
    u.q[1] = *(const float4*)(row + kb + 16 + hi);
    return u.v;
}

// B fragment (32x16 f16, N=lane&15) from transposed weights Wt[n][k]:
// same K order as A: lanes 0-15 hold K=kb+0..7,16..23; lanes 16-31 hold K=kb+8..15,24..31.
__device__ __forceinline__ v16h loadB(const _Float16* __restrict__ wcol, int lane, int kb) {
    const int ko = kb + ((lane & 16) ? 8 : 0);
    F16x16 u;
    u.q[0] = *(const float4*)(wcol + ko);
    u.q[1] = *(const float4*)(wcol + ko + 16);
    return u.v;
}

#define WMMA_F16(a, b, c) \
    __builtin_amdgcn_wmma_f32_16x16x32_f16(false, (a), false, (b), (short)0, (c), false, false)

// ---------------- edge bucketing (once; graph static across layers) ---------------
__global__ void k_init(int* hist) {
    if (threadIdx.x < NUM_RELS) hist[threadIdx.x] = 0;
}

__global__ void k_hist(const int* __restrict__ etypes, int* __restrict__ hist, int E) {
    int e = blockIdx.x * blockDim.x + threadIdx.x;
    if (e < E) atomicAdd(&hist[etypes[e]], 1);
}

// Segment starts padded to 16-edge alignment so every tile base is 16-element aligned.
__global__ void k_scan(const int* __restrict__ hist, int* __restrict__ segStart,
                       int* __restrict__ segEnd, int* __restrict__ tileOff,
                       int* __restrict__ cursor) {
    if (threadIdx.x == 0 && blockIdx.x == 0) {
        int pacc = 0, tacc = 0;
        for (int r = 0; r < NUM_RELS; ++r) {
            segStart[r] = pacc;
            segEnd[r]   = pacc + hist[r];
            cursor[r]   = pacc;
            tileOff[r]  = tacc;
            tacc += (hist[r] + 15) >> 4;
            pacc += (hist[r] + 15) & ~15;      // keep next segment 16-aligned
        }
        tileOff[NUM_RELS] = tacc;
    }
}

__global__ void k_scatter(const int* __restrict__ etypes, const int* __restrict__ src,
                          const int* __restrict__ dst, const float* __restrict__ norm,
                          int* __restrict__ cursor, int* __restrict__ srcS,
                          int* __restrict__ dstS, float* __restrict__ normS, int E) {
    int e = blockIdx.x * blockDim.x + threadIdx.x;
    if (e < E) {
        int r = etypes[e];
        int pos = atomicAdd(&cursor[r], 1);
        srcS[pos]  = src[e];
        dstS[pos]  = dst[e];
        normS[pos] = norm[e];
    }
}

__global__ void k_cast(const float* __restrict__ in, _Float16* __restrict__ out, int total4) {
    int i = blockIdx.x * blockDim.x + threadIdx.x;
    if (i < total4) {
        float4 x = ((const float4*)in)[i];
        _Float16* o = out + (size_t)i * 4;
        o[0] = (_Float16)x.x; o[1] = (_Float16)x.y;
        o[2] = (_Float16)x.z; o[3] = (_Float16)x.w;
    }
}

// -------- per-layer weight prep: Wt[r][n][k] = (comp.basis)^T (f16), loopT[n][k] ----------
__global__ void k_weights(const float* __restrict__ basis, const float* __restrict__ comp,
                          const float* __restrict__ loopw, _Float16* __restrict__ Wt,
                          _Float16* __restrict__ loopT, int dout, int doutP) {
    const int r = blockIdx.x;               // 0..15 relations, 16 -> self-loop weight
    if (r < NUM_RELS) {
        for (int idx = threadIdx.x; idx < doutP * DIN; idx += blockDim.x) {
            int n = idx / DIN, k = idx % DIN;
            float v = 0.f;
            if (n < dout) {
                #pragma unroll
                for (int b = 0; b < 8; ++b)
                    v += comp[r * 8 + b] * basis[((size_t)b * DIN + k) * dout + n];
            }
            Wt[((size_t)r * doutP + n) * DIN + k] = (_Float16)v;
        }
    } else {
        for (int idx = threadIdx.x; idx < doutP * DIN; idx += blockDim.x) {
            int n = idx / DIN, k = idx % DIN;
            loopT[(size_t)n * DIN + k] =
                (_Float16)((n < dout) ? loopw[(size_t)k * dout + n] : 0.f);
        }
    }
}

// -------- self-loop GEMM: agg = h @ loop_w + bias (initializes agg) --------
template <int DOUT, int DOUTP>
__global__ __launch_bounds__(32) void k_selfloop(
    const _Float16* __restrict__ hf16, const _Float16* __restrict__ loopT,
    const float* __restrict__ bias, float* __restrict__ agg, int n_nodes) {
    constexpr int NTL = DOUTP / 16;
    const int mt   = blockIdx.x / NTL;
    const int nt   = blockIdx.x % NTL;
    const int lane = threadIdx.x;
    const int m    = lane & 15;
    const int hi8  = (lane & 16) ? 8 : 0;
    const int row0 = mt * 16;

    int arowi = row0 + m; if (arowi >= n_nodes) arowi = n_nodes - 1;
    const _Float16* arow = hf16 + (size_t)arowi * DIN;
    v16h a0 = loadA(arow, lane, 0);
    v16h a1 = loadA(arow, lane, 32);

    const int n = nt * 16 + m;
    const _Float16* wcol = loopT + (size_t)n * DIN;
    v16h b0 = loadB(wcol, lane, 0);
    v16h b1 = loadB(wcol, lane, 32);

    float bn = (DOUT == DOUTP || n < DOUT) ? bias[n] : 0.f;
    v8f c = {bn, bn, bn, bn, bn, bn, bn, bn};
    c = WMMA_F16(a0, b0, c);
    c = WMMA_F16(a1, b1, c);

    float* outp = agg + ((size_t)(row0 + hi8) * DOUTP + n);
    if (row0 + 16 <= n_nodes) {                       // full tile: unguarded stores
        #pragma unroll
        for (int v = 0; v < 8; ++v) outp[(size_t)v * DOUTP] = c[v];
    } else {
        #pragma unroll
        for (int v = 0; v < 8; ++v)
            if (row0 + hi8 + v < n_nodes) outp[(size_t)v * DOUTP] = c[v];
    }
}

// -------- fused relation-GEMM + gather/scatter over relation-sorted edges --------
template <int DOUT, int DOUTP>
__global__ __launch_bounds__(32) void k_edges(
    const _Float16* __restrict__ hf16, const _Float16* __restrict__ Wt,
    const int* __restrict__ segStart, const int* __restrict__ segEnd,
    const int* __restrict__ tileOff,
    const int* __restrict__ srcS, const int* __restrict__ dstS,
    const float* __restrict__ normS, float* __restrict__ agg) {
    const int b = blockIdx.x;
    if (b >= tileOff[NUM_RELS]) return;
    int r = 0;
    while (b >= tileOff[r + 1]) ++r;               // uniform scalar search, <=16 iters
    const int base = segStart[r] + (b - tileOff[r]) * 16;   // 16-element aligned
    const int end  = segEnd[r];
    const int lane = threadIdx.x;
    const int m    = lane & 15;
    const int hi8  = (lane & 16) ? 8 : 0;
    const bool full = (base + 16 <= end);

    int pos = base + m; if (pos >= end) pos = end - 1;   // clamp tail rows (masked later)
    const _Float16* arow = hf16 + (size_t)srcS[pos] * DIN;
    v16h a0 = loadA(arow, lane, 0);
    v16h a1 = loadA(arow, lane, 32);

    int   dstv[8];
    float nrmv[8];
    if (full) {                                     // vector loads: base+hi8 is 8-aligned
        int4   d0 = *(const int4*)(dstS + base + hi8);
        int4   d1 = *(const int4*)(dstS + base + hi8 + 4);
        float4 n0 = *(const float4*)(normS + base + hi8);
        float4 n1 = *(const float4*)(normS + base + hi8 + 4);
        dstv[0] = d0.x; dstv[1] = d0.y; dstv[2] = d0.z; dstv[3] = d0.w;
        dstv[4] = d1.x; dstv[5] = d1.y; dstv[6] = d1.z; dstv[7] = d1.w;
        nrmv[0] = n0.x; nrmv[1] = n0.y; nrmv[2] = n0.z; nrmv[3] = n0.w;
        nrmv[4] = n1.x; nrmv[5] = n1.y; nrmv[6] = n1.z; nrmv[7] = n1.w;
    } else {
        #pragma unroll
        for (int v = 0; v < 8; ++v) {
            int e  = base + v + hi8;
            int ec = (e < end) ? e : end - 1;
            dstv[v] = dstS[ec];
            nrmv[v] = (e < end) ? normS[ec] : 0.f;
        }
    }

    const _Float16* wr = Wt + (size_t)r * DOUTP * DIN;
    constexpr int NTL = DOUTP / 16;
    #pragma unroll
    for (int nt = 0; nt < NTL; ++nt) {
        const int n = nt * 16 + m;
        const _Float16* wcol = wr + (size_t)n * DIN;
        v16h b0 = loadB(wcol, lane, 0);
        v16h b1 = loadB(wcol, lane, 32);
        v8f c = {};
        c = WMMA_F16(a0, b0, c);
        c = WMMA_F16(a1, b1, c);
        const bool ncol = (DOUT == DOUTP) || (n < DOUT);
        if (full) {
            if (ncol) {
                #pragma unroll
                for (int v = 0; v < 8; ++v)
                    unsafeAtomicAdd(agg + ((size_t)(unsigned)dstv[v] * DOUTP + n),
                                    c[v] * nrmv[v]);
            }
        } else if (ncol) {
            #pragma unroll
            for (int v = 0; v < 8; ++v)
                if (base + v + hi8 < end)
                    unsafeAtomicAdd(agg + ((size_t)(unsigned)dstv[v] * DOUTP + n),
                                    c[v] * nrmv[v]);
        }
    }
}

// ---------------- activations ----------------
__global__ void k_act_mid(const float* __restrict__ agg, _Float16* __restrict__ hf16,
                          int total4) {
    int i = blockIdx.x * blockDim.x + threadIdx.x;
    if (i < total4) {
        float4 x = ((const float4*)agg)[i];
        float4 y;
        y.x = x.x > 0.f ? x.x : 0.01f * x.x;
        y.y = x.y > 0.f ? x.y : 0.01f * x.y;
        y.z = x.z > 0.f ? x.z : 0.01f * x.z;
        y.w = x.w > 0.f ? x.w : 0.01f * x.w;
        _Float16* o = hf16 + (size_t)i * 4;
        o[0] = (_Float16)y.x; o[1] = (_Float16)y.y;
        o[2] = (_Float16)y.z; o[3] = (_Float16)y.w;
    }
}

__global__ void k_act_last(const float* __restrict__ agg, float* __restrict__ out,
                           int n_nodes) {
    int i = blockIdx.x * blockDim.x + threadIdx.x;   // one thread per node
    if (i < n_nodes) {
        float4 a = *(const float4*)(agg + (size_t)i * 16);     // doutP = 16 stride
        float4 b = *(const float4*)(agg + (size_t)i * 16 + 4);
        a.x = fmaxf(a.x, 0.f); a.y = fmaxf(a.y, 0.f);
        a.z = fmaxf(a.z, 0.f); a.w = fmaxf(a.w, 0.f);
        b.x = fmaxf(b.x, 0.f); b.y = fmaxf(b.y, 0.f);
        b.z = fmaxf(b.z, 0.f); b.w = fmaxf(b.w, 0.f);
        *(float4*)(out + (size_t)i * 8)     = a;
        *(float4*)(out + (size_t)i * 8 + 4) = b;
    }
}

extern "C" void kernel_launch(void* const* d_in, const int* in_sizes, int n_in,
                              void* d_out, int out_size, void* d_ws, size_t ws_size,
                              hipStream_t stream) {
    const float* features = (const float*)d_in[0];
    const float* norm     = (const float*)d_in[1];
    const float* basis[3] = {(const float*)d_in[2], (const float*)d_in[6],  (const float*)d_in[10]};
    const float* comp[3]  = {(const float*)d_in[3], (const float*)d_in[7],  (const float*)d_in[11]};
    const float* loopw[3] = {(const float*)d_in[4], (const float*)d_in[8],  (const float*)d_in[12]};
    const float* bias[3]  = {(const float*)d_in[5], (const float*)d_in[9],  (const float*)d_in[13]};
    const int* etypes = (const int*)d_in[14];
    const int* src    = (const int*)d_in[15];
    const int* dst    = (const int*)d_in[16];

    const int E  = in_sizes[14];
    const int N  = in_sizes[0] / DIN;
    const int EP = E + NUM_RELS * 16;     // room for 16-alignment padding per segment

    // workspace carve-out (256B aligned regions)
    char* ws = (char*)d_ws;
    size_t off = 0;
    auto carve = [&](size_t bytes) -> char* {
        char* p = ws + off;
        off = (off + bytes + 255) & ~(size_t)255;
        return p;
    };
    int*       hist     = (int*)carve(NUM_RELS * 4);
    int*       segStart = (int*)carve(NUM_RELS * 4);
    int*       segEnd   = (int*)carve(NUM_RELS * 4);
    int*       tileOff  = (int*)carve((NUM_RELS + 1) * 4);
    int*       cursor   = (int*)carve(NUM_RELS * 4);
    int*       srcS     = (int*)carve((size_t)EP * 4);
    int*       dstS     = (int*)carve((size_t)EP * 4);
    float*     normS    = (float*)carve((size_t)EP * 4);
    _Float16*  hf16     = (_Float16*)carve((size_t)N * DIN * 2);
    float*     agg      = (float*)carve((size_t)N * DIN * 4);
    _Float16*  Wt       = (_Float16*)carve((size_t)NUM_RELS * DIN * DIN * 2);
    _Float16*  loopT    = (_Float16*)carve((size_t)DIN * DIN * 2);

    // --- one-time edge bucketing by relation ---
    k_init<<<1, 64, 0, stream>>>(hist);
    k_hist<<<(E + 255) / 256, 256, 0, stream>>>(etypes, hist, E);
    k_scan<<<1, 1, 0, stream>>>(hist, segStart, segEnd, tileOff, cursor);
    k_scatter<<<(E + 255) / 256, 256, 0, stream>>>(etypes, src, dst, norm,
                                                   cursor, srcS, dstS, normS, E);
    k_cast<<<(N * DIN / 4 + 255) / 256, 256, 0, stream>>>(features, hf16, N * DIN / 4);

    const int mtiles = (N + 15) / 16;
    const int etiles = E / 16 + NUM_RELS;   // upper bound on total tile count

    for (int l = 0; l < 3; ++l) {
        const int dout  = (l < 2) ? 64 : 8;
        const int doutP = (l < 2) ? 64 : 16;
        k_weights<<<NUM_RELS + 1, 256, 0, stream>>>(basis[l], comp[l], loopw[l],
                                                    Wt, loopT, dout, doutP);
        if (l < 2) {
            k_selfloop<64, 64><<<mtiles * 4, 32, 0, stream>>>(hf16, loopT, bias[l], agg, N);
            k_edges<64, 64><<<etiles, 32, 0, stream>>>(hf16, Wt, segStart, segEnd, tileOff,
                                                       srcS, dstS, normS, agg);
            k_act_mid<<<(N * DIN / 4 + 255) / 256, 256, 0, stream>>>(agg, hf16, N * DIN / 4);
        } else {
            k_selfloop<8, 16><<<mtiles * 1, 32, 0, stream>>>(hf16, loopT, bias[l], agg, N);
            k_edges<8, 16><<<etiles, 32, 0, stream>>>(hf16, Wt, segStart, segEnd, tileOff,
                                                      srcS, dstS, normS, agg);
            k_act_last<<<(N + 255) / 256, 256, 0, stream>>>(agg, (float*)d_out, N);
        }
    }
}
